// ClusterOverlap_5308579578516
// MI455X (gfx1250) — matrix operand order repaired and checked by reading, not compiled, not run on hardware
//
#include <hip/hip_runtime.h>
#include <hip/hip_bf16.h>
#include <math.h>

#define Bn     8192
#define ENCD   256
#define NCL    32
#define LIST   26   // k+1 = 26 smallest kept per row

typedef __attribute__((ext_vector_type(16))) __bf16 v16bf;
typedef __attribute__((ext_vector_type(8)))  __bf16 bf16x8;
typedef __attribute__((ext_vector_type(8)))  float  v8f;

union AFrag { v16bf v; bf16x8 h[2]; };

__device__ __forceinline__ unsigned long long sx64(unsigned long long v, int m) {
    unsigned lo = (unsigned)__shfl_xor((int)(unsigned)v, m, 32);
    unsigned hi = (unsigned)__shfl_xor((int)(unsigned)(v >> 32), m, 32);
    return ((unsigned long long)hi << 32) | lo;
}

// --- CDNA5 async global->LDS copy: offset immediate applies to BOTH addresses ---
#define ALD(l, g, o) asm volatile("global_load_async_to_lds_b128 %0, %1, off offset:" o \
                                  :: "v"(l), "v"(g) : "memory")
__device__ __forceinline__ void async_copy128B(unsigned l, unsigned long long g) {
    ALD(l, g, "0");  ALD(l, g, "16"); ALD(l, g, "32");  ALD(l, g, "48");
    ALD(l, g, "64"); ALD(l, g, "80"); ALD(l, g, "96");  ALD(l, g, "112");
}
__device__ __forceinline__ void wait_async0() {
    asm volatile("s_wait_asynccnt 0" ::: "memory");
}

// ---------- prep 1: split fp32 encodings into hi/lo bf16 planes ----------
__global__ void prep_split(const float* __restrict__ enc,
                           __bf16* __restrict__ hi, __bf16* __restrict__ lo) {
    int idx = blockIdx.x * 256 + threadIdx.x;
    float x = enc[idx];
    __bf16 h = (__bf16)x;
    hi[idx] = h;
    lo[idx] = (__bf16)(x - (float)h);
}

// ---------- prep 2: per-row sq-norm, argmax cluster, max confidence ----------
__global__ void prep_rows(const float* __restrict__ enc, const float* __restrict__ cat,
                          float* __restrict__ sq, unsigned* __restrict__ cid,
                          float* __restrict__ maxg) {
    int lane = threadIdx.x & 31;
    int row  = blockIdx.x * 8 + (threadIdx.x >> 5);
    float s = 0.f;
#pragma unroll
    for (int t = 0; t < 8; t++) { float x = enc[(long)row * ENCD + t * 32 + lane]; s += x * x; }
#pragma unroll
    for (int off = 16; off >= 1; off >>= 1) s += __shfl_xor(s, off, 32);

    float c = cat[row * NCL + lane];   // softmax output, positive -> bits are order-preserving
    unsigned long long key = (((unsigned long long)__float_as_uint(c)) << 6) | (unsigned)(63 - lane);
#pragma unroll
    for (int off = 16; off >= 1; off >>= 1) {
        unsigned long long o = sx64(key, off); key = o > key ? o : key;
    }
    if (lane == 0) {
        sq[row]   = s;
        cid[row]  = (unsigned)(63 - (int)(key & 63ull));       // first-index tie-break like argmax
        maxg[row] = __uint_as_float((unsigned)(key >> 6));
    }
}

// ---------- main fused kernel: split-bf16 WMMA cdist + top-26 select + entropy ----------
__launch_bounds__(256, 1)
__global__ void cluster_overlap_main(const __bf16* __restrict__ ehi, const __bf16* __restrict__ elo,
                                     const float* __restrict__ sqg, const unsigned* __restrict__ cidg,
                                     const float* __restrict__ maxg, float* __restrict__ out) {
    __shared__ __bf16   Bsh[2][2][32][264];  // [buffer][plane][col][K], padded stride
    __shared__ float    distb[64][33];
    __shared__ float    listd[64][LIST];
    __shared__ unsigned listc[64][LIST];
    __shared__ float    sqr[64];
    __shared__ unsigned hist[8][32];

    const int tid = threadIdx.x;
    const int w = tid >> 5, lane = tid & 31;
    const int rowPanel = blockIdx.x * 64;

    for (int i = tid; i < 64 * LIST; i += 256) {
        (&listd[0][0])[i] = INFINITY;
        (&listc[0][0])[i] = 0u;
    }
    if (tid < 64) sqr[tid] = sqg[rowPanel + tid];

    // A fragments resident in VGPRs: wave's 16-row tile, all K, hi+lo (128 VGPRs)
    const int rt = w >> 1, ct = w & 1;
    const int m = lane & 15, hh = lane >> 4;
    const long arow = rowPanel + rt * 16 + m;
    AFrag ahi[8], alo[8];
#pragma unroll
    for (int s = 0; s < 8; s++) {
        const int k0 = 32 * s;   // CDNA5 bf16 A 16x32 layout: lo-half K = k0+8h..+7, hi-half K = k0+16+8h..+7
        ahi[s].h[0] = *(const bf16x8*)(ehi + arow * ENCD + k0 + 8 * hh);
        ahi[s].h[1] = *(const bf16x8*)(ehi + arow * ENCD + k0 + 16 + 8 * hh);
        alo[s].h[0] = *(const bf16x8*)(elo + arow * ENCD + k0 + 8 * hh);
        alo[s].h[1] = *(const bf16x8*)(elo + arow * ENCD + k0 + 16 + 8 * hh);
    }

    // each thread owns a 128B slice of the 32KB chunk copy (2 planes x 32 cols x 512B)
    const int rc = tid >> 2, sub = tid & 3;
    const int pp = rc >> 5, rowc = rc & 31;
    const __bf16* gsrcBase = (pp ? elo : ehi) + (long)rowc * ENCD + sub * 64;
    const unsigned lds0 = (unsigned)(size_t)&Bsh[0][pp][rowc][sub * 64];
    const unsigned lds1 = (unsigned)(size_t)&Bsh[1][pp][rowc][sub * 64];

    // prefetch chunk 0 into buffer 0
    async_copy128B(lds0, (unsigned long long)(size_t)gsrcBase);

    const int bcol = ct * 16 + m;       // B 32x16 layout: lane = column, lo-half K = k0+8h..+7, hi-half K = k0+16+8h..+7
    const int cl = ct * 16 + m;

    for (int cidx = 0; cidx < Bn / 32; cidx++) {
        const int buf = cidx & 1;
        wait_async0();        // my slice of chunk cidx landed in LDS
        __syncthreads();      // everyone's slice landed; prev scan done -> distb & other buffer free

        if (cidx + 1 < Bn / 32) {   // overlap: fetch next chunk into the other buffer
            unsigned long long g =
                (unsigned long long)(size_t)(gsrcBase + (long)(cidx + 1) * 32 * ENCD);
            async_copy128B(buf ? lds0 : lds1, g);
        }

        // 16x16 tile per wave: three independent accumulator chains (hi*hi, hi*lo, lo*hi)
        v8f a0 = {0.f, 0.f, 0.f, 0.f, 0.f, 0.f, 0.f, 0.f};
        v8f a1 = a0, a2 = a0;
#pragma unroll
        for (int s = 0; s < 8; s++) {
            const int k0 = 32 * s;
            AFrag bhi, blo;
            bhi.h[0] = *(const bf16x8*)&Bsh[buf][0][bcol][k0 + 8 * hh];
            bhi.h[1] = *(const bf16x8*)&Bsh[buf][0][bcol][k0 + 16 + 8 * hh];
            blo.h[0] = *(const bf16x8*)&Bsh[buf][1][bcol][k0 + 8 * hh];
            blo.h[1] = *(const bf16x8*)&Bsh[buf][1][bcol][k0 + 16 + 8 * hh];
            a0 = __builtin_amdgcn_wmma_f32_16x16x32_bf16(false, ahi[s].v, false, bhi.v, (short)0, a0, false, false);
            a1 = __builtin_amdgcn_wmma_f32_16x16x32_bf16(false, ahi[s].v, false, blo.v, (short)0, a1, false, false);
            a2 = __builtin_amdgcn_wmma_f32_16x16x32_bf16(false, alo[s].v, false, bhi.v, (short)0, a2, false, false);
        }

        // d2 -> dist, into LDS (C layout: row = v + 8*(lane>=16), col = lane&15)
        const int c0 = cidx * 32;
        const float sqcv = sqg[c0 + cl];
#pragma unroll
        for (int v = 0; v < 8; v++) {
            int rl = rt * 16 + v + 8 * hh;
            float dot = a0[v] + a1[v] + a2[v];
            float d2 = sqr[rl] + sqcv - 2.f * dot;
            distb[rl][cl] = sqrtf(fmaxf(d2, 0.f));
        }
        __syncthreads();      // dist tile complete for all waves

        // selection scan: wave w owns rows [8w, 8w+8); keep 26 smallest (dist, cluster)
        const unsigned cv = cidg[c0 + lane];
        for (int rr = 0; rr < 8; rr++) {
            int r = w * 8 + rr;
            float val = distb[r][lane];
            float dl = (lane < LIST) ? listd[r][lane] : 0.f;   // entries >= 0, bits order-preserving
            unsigned long long key = (lane < LIST)
                ? ((((unsigned long long)__float_as_uint(dl)) << 6) | (unsigned)lane) : 0ull;
#pragma unroll
            for (int off = 16; off >= 1; off >>= 1) {
                unsigned long long o = sx64(key, off); key = o > key ? o : key;
            }
            float thr0 = __uint_as_float((unsigned)(key >> 6));
            unsigned mask = (unsigned)__ballot(val < thr0);
            while (mask) {
                int b = __ffs(mask) - 1; mask &= mask - 1;
                float vb = __shfl(val, b, 32);
                unsigned cb = (unsigned)__shfl((int)cv, b, 32);
                float dl2 = (lane < LIST) ? listd[r][lane] : 0.f;
                unsigned long long k2 = (lane < LIST)
                    ? ((((unsigned long long)__float_as_uint(dl2)) << 6) | (unsigned)lane) : 0ull;
#pragma unroll
                for (int off = 16; off >= 1; off >>= 1) {
                    unsigned long long o = sx64(k2, off); k2 = o > k2 ? o : k2;
                }
                float mx = __uint_as_float((unsigned)(k2 >> 6));
                int am = (int)(k2 & 63ull);
                if (vb < mx && lane == am) { listd[r][am] = vb; listc[r][am] = cb; }
            }
        }
        // no barrier here: next iteration's top barrier separates scan from distb overwrite
    }

    __syncthreads();
    // epilogue: thr = max of list = sorted[k]; neighbors are entries < thr; entropy
    for (int rr = 0; rr < 8; rr++) {
        int r = w * 8 + rr;
        float dl = (lane < LIST) ? listd[r][lane] : 0.f;
        unsigned cl_ = (lane < LIST) ? listc[r][lane] : 0u;
        unsigned long long key = (lane < LIST)
            ? ((((unsigned long long)__float_as_uint(dl)) << 6) | (unsigned)lane) : 0ull;
#pragma unroll
        for (int off = 16; off >= 1; off >>= 1) {
            unsigned long long o = sx64(key, off); key = o > key ? o : key;
        }
        float thr = __uint_as_float((unsigned)(key >> 6));
        bool pred = (lane < LIST) && (dl < thr);
        int n = __popc((unsigned)__ballot(pred));
        hist[w][lane] = 0u;
        if (pred) atomicAdd(&hist[w][cl_], 1u);
        unsigned cnt = hist[w][lane];                 // lane == cluster id
        float bfrac = (float)cnt / (float)n;
        float e = -bfrac * logf(bfrac + 1e-5f);
#pragma unroll
        for (int off = 16; off >= 1; off >>= 1) e += __shfl_xor(e, off, 32);
        if (lane == 0) {
            int grow = rowPanel + r;
            out[grow] = e * maxg[grow];
        }
    }
}

extern "C" void kernel_launch(void* const* d_in, const int* in_sizes, int n_in,
                              void* d_out, int out_size, void* d_ws, size_t ws_size,
                              hipStream_t stream) {
    const float* enc = (const float*)d_in[0];
    const float* cat = (const float*)d_in[1];
    (void)in_sizes; (void)n_in; (void)out_size; (void)ws_size;  // k == 25 (compile-time LIST)

    char* ws = (char*)d_ws;
    const size_t planeB = (size_t)Bn * ENCD * sizeof(__bf16);   // 4 MB
    __bf16*   ehi = (__bf16*)ws;
    __bf16*   elo = (__bf16*)(ws + planeB);
    float*    sq  = (float*)(ws + 2 * planeB);
    float*    mgx = (float*)(ws + 2 * planeB + (size_t)Bn * 4);
    unsigned* cid = (unsigned*)(ws + 2 * planeB + (size_t)Bn * 8);
    float*    out = (float*)d_out;

    prep_split<<<(Bn * ENCD) / 256, 256, 0, stream>>>(enc, ehi, elo);
    prep_rows<<<Bn / 8, 256, 0, stream>>>(enc, cat, sq, cid, mgx);
    cluster_overlap_main<<<Bn / 64, 256, 0, stream>>>(ehi, elo, sq, cid, mgx, out);
}
